// HilbertAttentionTritonSimple_58926951301482
// MI455X (gfx1250) — compile-verified, matrix operand order checked by reading, not run on hardware
//
#include <hip/hip_runtime.h>
#include <hip/hip_bf16.h>

// ---------------- problem constants ----------------
#define B_    2
#define M_    8192
#define D_    1024
#define H_    16
#define HD_   64
#define SEG_  128
#define DIL_  2
#define NSEG_ (M_ / SEG_)   // 64
#define NKEY_ (SEG_ / DIL_) // 64
#define E3_   (3 * D_)      // 3072
#define ROWS_ (B_ * M_)     // 16384
#define KT_   64            // GEMM k-tile staged via TDM

typedef __attribute__((ext_vector_type(16))) __bf16 bf16x16;
typedef __attribute__((ext_vector_type(8)))  float  f32x8;
typedef __attribute__((ext_vector_type(4)))  unsigned u32x4;
typedef __attribute__((ext_vector_type(8)))  int      i32x8;
typedef __attribute__((ext_vector_type(4)))  int      i32x4;

// fp32 -> bf16, round-to-nearest-even
static __device__ __forceinline__ __bf16 f2bf(float f) {
    unsigned u = __builtin_bit_cast(unsigned, f);
    unsigned r = u + 0x7FFFu + ((u >> 16) & 1u);
    return __builtin_bit_cast(__bf16, (unsigned short)(r >> 16));
}

// CDNA5 16-bit A/B fragment K mapping (ISA 7.12.2):
// lanes 0-15: vgpr 0..3 -> K 0..7, vgpr 4..7 -> K 16..23
// lanes 16-31: vgpr 0..3 -> K 8..15, vgpr 4..7 -> K 24..31
static __device__ __forceinline__ int kmap_base(int i, int half) {
    return ((i & 4) << 2) + half * 8 + ((i & 3) << 1);
}

// A-style fragment: element (row=lane&15, k) from row-major [16 x >=32] (also
// serves as B fragment when B's source is stored N-major, i.e. row = N).
static __device__ __forceinline__ bf16x16 load_frag_row(const __bf16* p, int ld) {
    const int lane = threadIdx.x & 31;
    const int half = lane >> 4, r = lane & 15;
    bf16x16 f;
#pragma unroll
    for (int i = 0; i < 8; ++i) {
        unsigned u = *(const unsigned*)(p + r * ld + kmap_base(i, half));
        f[2 * i]     = __builtin_bit_cast(__bf16, (unsigned short)(u & 0xFFFFu));
        f[2 * i + 1] = __builtin_bit_cast(__bf16, (unsigned short)(u >> 16));
    }
    return f;
}

// B fragment from K-major storage: element (k, n=lane&15) from src[k][n], ld over n.
static __device__ __forceinline__ bf16x16 load_frag_col(const __bf16* p, int ld) {
    const int lane = threadIdx.x & 31;
    const int half = lane >> 4, r = lane & 15;
    bf16x16 f;
#pragma unroll
    for (int i = 0; i < 8; ++i) {
        int kb = kmap_base(i, half);
        f[2 * i]     = p[kb * ld + r];
        f[2 * i + 1] = p[(kb + 1) * ld + r];
    }
    return f;
}

static __device__ __forceinline__ f32x8 wmma_bf16(bf16x16 a, bf16x16 b, f32x8 c) {
    return __builtin_amdgcn_wmma_f32_16x16x32_bf16(false, a, false, b, (short)0, c,
                                                   false, false);
}

// One 16x64 (rows x cols) output sub-tile over a K=64 slice:
// loads ALL fragments first (one dscnt wait), then 8 WMMAs in 4 independent
// accumulator chains so the XDL pipeline stays busy.
static __device__ __forceinline__ void mma_16rows_4cols_k64(const __bf16* aP, int lda,
                                                            const __bf16* bP, int ldb,
                                                            f32x8 acc[4]) {
    bf16x16 a0 = load_frag_row(aP, lda);
    bf16x16 a1 = load_frag_row(aP + 32, lda);
    bf16x16 b0[4], b1[4];
#pragma unroll
    for (int c = 0; c < 4; ++c) {
        b0[c] = load_frag_row(bP + c * 16 * ldb, ldb);
        b1[c] = load_frag_row(bP + c * 16 * ldb + 32, ldb);
    }
#pragma unroll
    for (int c = 0; c < 4; ++c)
        acc[c] = wmma_bf16(a0, b0[c], acc[c]);
#pragma unroll
    for (int c = 0; c < 4; ++c)
        acc[c] = wmma_bf16(a1, b1[c], acc[c]);
}

// Generic-pointer low 32 bits == LDS byte offset (ISA aperture mapping).
static __device__ __forceinline__ unsigned lds_off(const void* p) {
    return (unsigned)(unsigned long long)(uintptr_t)p;
}

// TDM 2D tile load: global (row-major, elem stride ld_elems, bf16) -> LDS
// contiguous [tile_d1][tile_d0]. Descriptor per CDNA5 ISA ch.8 (D# groups).
static __device__ __forceinline__ void tdm_load_tile2d(unsigned ldsDst,
                                                       const __bf16* gsrc,
                                                       unsigned ld_elems,
                                                       unsigned tile_d0,
                                                       unsigned tile_d1) {
    unsigned long long ga = (unsigned long long)(uintptr_t)gsrc;
    u32x4 g0;
    g0[0] = 1u;                                  // count=1, user mode, no gather
    g0[1] = ldsDst;                              // lds_addr (bytes)
    g0[2] = (unsigned)ga;                        // global_addr[31:0]
    g0[3] = (unsigned)(ga >> 32) | (2u << 30);   // global_addr[56:32] | type=2
    const unsigned td0 = ld_elems;               // extent from tile start (>= tile_d0)
    const unsigned td1 = tile_d1;                // rows exactly
    i32x8 g1;
    g1[0] = 1 << 16;                                           // data_size=2B
    g1[1] = (int)((td0 & 0xFFFFu) << 16);                      // tensor_dim0 lo
    g1[2] = (int)((td0 >> 16) | ((td1 & 0xFFFFu) << 16));      // dim0 hi | dim1 lo
    g1[3] = (int)((td1 >> 16) | (tile_d0 << 16));              // dim1 hi | tile_dim0
    g1[4] = (int)tile_d1;                                      // tile_dim1 (tile_dim2=0)
    g1[5] = (int)ld_elems;                                     // tensor_dim0_stride lo
    g1[6] = 0;                                                 // stride hi | dim1_stride lo
    g1[7] = 0;
    i32x4 z4 = {0, 0, 0, 0};
#if defined(__clang_major__) && __clang_major__ >= 23
    i32x8 z8 = {0, 0, 0, 0, 0, 0, 0, 0};
    __builtin_amdgcn_tensor_load_to_lds(g0, g1, z4, z4, z8, 0);
#else
    __builtin_amdgcn_tensor_load_to_lds(g0, g1, z4, z4, 0);
#endif
}

// ---------------- fp32 -> bf16 conversion ----------------
__global__ void cvt_bf16_kernel(const float* __restrict__ src,
                                __bf16* __restrict__ dst, int n) {
    int i = blockIdx.x * blockDim.x + threadIdx.x;
    if (i < n) dst[i] = f2bf(src[i]);
}

// TDM double-buffered 128x64 GEMM tile body: acc[c] += A[128,K] * W[N,K]^T.
// Block: 256 threads (8 waves); wave w owns rows [w*16, w*16+16) of the tile.
static __device__ __forceinline__ void gemm_body_tdm(const __bf16* __restrict__ A,
                                                     const __bf16* __restrict__ W,
                                                     int row0, int col0, int K,
                                                     f32x8 acc[4]) {
    __shared__ __align__(128) __bf16 sA[2][128 * KT_];
    __shared__ __align__(128) __bf16 sB[2][64 * KT_];
    const int wave = threadIdx.x >> 5;
    const int NKT = K / KT_;

    if (wave == 0) {
        tdm_load_tile2d(lds_off(&sA[0][0]), A + (size_t)row0 * K, K, KT_, 128);
        tdm_load_tile2d(lds_off(&sB[0][0]), W + (size_t)col0 * K, K, KT_, 64);
    }
    for (int kt = 0; kt < NKT; ++kt) {
        const int cur = kt & 1;
        if (wave == 0) {
            if (kt + 1 < NKT) {
                const int k0 = (kt + 1) * KT_;
                tdm_load_tile2d(lds_off(&sA[cur ^ 1][0]),
                                A + (size_t)row0 * K + k0, K, KT_, 128);
                tdm_load_tile2d(lds_off(&sB[cur ^ 1][0]),
                                W + (size_t)col0 * K + k0, K, KT_, 64);
                __builtin_amdgcn_s_wait_tensorcnt(2);  // current stage's 2 done
            } else {
                __builtin_amdgcn_s_wait_tensorcnt(0);
            }
        }
        __syncthreads();  // staged tiles visible to all waves
        mma_16rows_4cols_k64(&sA[cur][wave * 16 * KT_], KT_, &sB[cur][0], KT_, acc);
        __syncthreads();  // all reads done before buffer is re-filled
    }
}

// ---------------- QKV GEMM ----------------
// qkv[row, e] = sum_d x[row, d] * Wqkv[e, d]; scatter into q/k/v [B,H,M,64] bf16.
__global__ __launch_bounds__(256)
void qkv_gemm_kernel(const __bf16* __restrict__ xb, const __bf16* __restrict__ wb,
                     __bf16* __restrict__ qb, __bf16* __restrict__ kb,
                     __bf16* __restrict__ vb) {
    const int col0 = blockIdx.x * 64;   // over E3_
    const int row0 = blockIdx.y * 128;  // over ROWS_
    const int wave = threadIdx.x >> 5;
    const int lane = threadIdx.x & 31;
    const int half = lane >> 4, r = lane & 15;
    const int rowT = row0 + wave * 16;

    f32x8 acc[4] = {};
    gemm_body_tdm(xb, wb, row0, col0, D_, acc);

#pragma unroll
    for (int c = 0; c < 4; ++c) {
        int e = col0 + c * 16 + r;
        int which = e >> 10;
        int h = (e >> 6) & (H_ - 1);
        int hd = e & (HD_ - 1);
        __bf16* dstb = (which == 0) ? qb : ((which == 1) ? kb : vb);
        float scl = (which == 0) ? 0.125f : 1.0f;  // hd^-0.5 on q
#pragma unroll
        for (int i = 0; i < 8; ++i) {
            int row = rowT + half * 8 + i;
            int bb = row >> 13, m = row & (M_ - 1);
            size_t dst = (((size_t)bb * H_ + h) * M_ + m) * HD_ + hd;
            dstb[dst] = f2bf(acc[c][i] * scl);
        }
    }
}

// ---------------- output GEMM ----------------
// out[row, n] = sum_d att[row, d] * Wout[n, d]  (fp32 out)
__global__ __launch_bounds__(256)
void out_gemm_kernel(const __bf16* __restrict__ ab, const __bf16* __restrict__ wb,
                     float* __restrict__ out) {
    const int col0 = blockIdx.x * 64;   // over D_
    const int row0 = blockIdx.y * 128;  // over ROWS_
    const int wave = threadIdx.x >> 5;
    const int lane = threadIdx.x & 31;
    const int half = lane >> 4, r = lane & 15;
    const int rowT = row0 + wave * 16;

    f32x8 acc[4] = {};
    gemm_body_tdm(ab, wb, row0, col0, D_, acc);

#pragma unroll
    for (int c = 0; c < 4; ++c) {
        int n = col0 + c * 16 + r;
#pragma unroll
        for (int i = 0; i < 8; ++i) {
            int row = rowT + half * 8 + i;
            out[(size_t)row * D_ + n] = acc[c][i];
        }
    }
}

// ---------------- attention (per b,h,segment) ----------------
__global__ __launch_bounds__(128)
void attn_kernel(const __bf16* __restrict__ qb, const __bf16* __restrict__ kb,
                 const __bf16* __restrict__ vb, const int* __restrict__ hmap,
                 __bf16* __restrict__ attb) {
    __shared__ __align__(16) __bf16 sK[NKEY_ * HD_];
    __shared__ __align__(16) __bf16 sV[NKEY_ * HD_];
    __shared__ __align__(16) __bf16 sW[SEG_ * NKEY_];
    __shared__ float sSum[SEG_];
    __shared__ int   sIdx[NKEY_];

    const int bid = blockIdx.x;
    const int seg = bid % NSEG_;
    const int h   = (bid / NSEG_) % H_;
    const int b   = bid / (NSEG_ * H_);
    const int tid = threadIdx.x;
    const int wave = tid >> 5, lane = tid & 31;
    const int half = lane >> 4, r = lane & 15;

    if (tid < NKEY_) sIdx[tid] = hmap[seg * SEG_ + tid * DIL_];
    __syncthreads();

    {   // gather K,V rows via hilbert map: thread t -> row t/2, 32-col half
        int rowk = tid >> 1;
        int c0 = (tid & 1) * 32;
        int phys = sIdx[rowk];
        const __bf16* ksrc = kb + (((size_t)b * H_ + h) * M_ + phys) * HD_ + c0;
        const __bf16* vsrc = vb + (((size_t)b * H_ + h) * M_ + phys) * HD_ + c0;
#pragma unroll
        for (int j = 0; j < 32; j += 8) {
            *(uint4*)&sK[rowk * HD_ + c0 + j] = *(const uint4*)(ksrc + j);
            *(uint4*)&sV[rowk * HD_ + c0 + j] = *(const uint4*)(vsrc + j);
        }
    }
    __syncthreads();

    const __bf16* qbase = qb + (((size_t)b * H_ + h) * M_ + seg * SEG_) * HD_;

    // phase 1: scores = Q.K^T, w = exp(scores), row sums, w -> LDS (bf16)
    for (int t = 0; t < 2; ++t) {
        int qt = wave + t * 4;  // 8 row-tiles of 16
        f32x8 s[4] = {};
        mma_16rows_4cols_k64(qbase + qt * 16 * HD_, HD_, sK, HD_, s);
#pragma unroll
        for (int i = 0; i < 8; ++i) {
            float rs = 0.f;
#pragma unroll
            for (int c = 0; c < 4; ++c) {
                float e = __expf(s[c][i]);
                s[c][i] = e;
                rs += e;
            }
            rs += __shfl_xor(rs, 1, 32);
            rs += __shfl_xor(rs, 2, 32);
            rs += __shfl_xor(rs, 4, 32);
            rs += __shfl_xor(rs, 8, 32);  // sum over 16 lanes of this half
            int row = qt * 16 + half * 8 + i;
            if (r == 0) sSum[row] = rs;
#pragma unroll
            for (int c = 0; c < 4; ++c)
                sW[row * NKEY_ + c * 16 + r] = f2bf(s[c][i]);
        }
    }
    __syncthreads();

    // phase 2: out = (w @ V) / (rowsum + 1e-6), store bf16 [B*M, D]
    for (int t = 0; t < 2; ++t) {
        int qt = wave + t * 4;
        f32x8 o[4] = {};
        {   // batch all fragments, then 8 WMMAs (B from K-major sV)
            const __bf16* aP = sW + qt * 16 * NKEY_;
            bf16x16 a0 = load_frag_row(aP, NKEY_);
            bf16x16 a1 = load_frag_row(aP + 32, NKEY_);
            bf16x16 b0[4], b1[4];
#pragma unroll
            for (int c = 0; c < 4; ++c) {
                b0[c] = load_frag_col(sV + c * 16, HD_);
                b1[c] = load_frag_col(sV + 32 * HD_ + c * 16, HD_);
            }
#pragma unroll
            for (int c = 0; c < 4; ++c)
                o[c] = wmma_bf16(a0, b0[c], o[c]);
#pragma unroll
            for (int c = 0; c < 4; ++c)
                o[c] = wmma_bf16(a1, b1[c], o[c]);
        }
#pragma unroll
        for (int i = 0; i < 8; ++i) {
            int row = qt * 16 + half * 8 + i;
            float inv = 1.0f / (sSum[row] + 1e-6f);
            int m = seg * SEG_ + row;
            size_t dst = ((size_t)b * M_ + m) * D_ + h * HD_;
#pragma unroll
            for (int c = 0; c < 4; ++c)
                attb[dst + c * 16 + r] = f2bf(o[c][i] * inv);
        }
    }
}

// ---------------- launcher ----------------
extern "C" void kernel_launch(void* const* d_in, const int* in_sizes, int n_in,
                              void* d_out, int out_size, void* d_ws, size_t ws_size,
                              hipStream_t stream) {
    (void)in_sizes; (void)n_in; (void)out_size; (void)ws_size;
    const float* x    = (const float*)d_in[0];
    const float* Wqkv = (const float*)d_in[1];
    const float* Wout = (const float*)d_in[2];
    const int*   hmap = (const int*)d_in[3];
    float* out = (float*)d_out;

    char* ws = (char*)d_ws;
    size_t off = 0;
    auto alloc = [&](size_t bytes) -> void* {
        off = (off + 255) & ~(size_t)255;
        void* p = ws + off;
        off += bytes;
        return p;
    };
    __bf16* xb    = (__bf16*)alloc((size_t)ROWS_ * D_ * 2);
    __bf16* wqkvb = (__bf16*)alloc((size_t)E3_ * D_ * 2);
    __bf16* woutb = (__bf16*)alloc((size_t)D_ * D_ * 2);
    __bf16* qb    = (__bf16*)alloc((size_t)B_ * H_ * M_ * HD_ * 2);
    __bf16* kb    = (__bf16*)alloc((size_t)B_ * H_ * M_ * HD_ * 2);
    __bf16* vb    = (__bf16*)alloc((size_t)B_ * H_ * M_ * HD_ * 2);
    __bf16* attb  = (__bf16*)alloc((size_t)ROWS_ * D_ * 2);

    const int nx = ROWS_ * D_;
    const int nq = E3_ * D_;
    const int no = D_ * D_;
    cvt_bf16_kernel<<<(nx + 255) / 256, 256, 0, stream>>>(x, xb, nx);
    cvt_bf16_kernel<<<(nq + 255) / 256, 256, 0, stream>>>(Wqkv, wqkvb, nq);
    cvt_bf16_kernel<<<(no + 255) / 256, 256, 0, stream>>>(Wout, woutb, no);

    qkv_gemm_kernel<<<dim3(E3_ / 64, ROWS_ / 128), 256, 0, stream>>>(xb, wqkvb, qb,
                                                                     kb, vb);
    attn_kernel<<<B_ * H_ * NSEG_, 128, 0, stream>>>(qb, kb, vb, hmap, attb);
    out_gemm_kernel<<<dim3(D_ / 64, ROWS_ / 128), 256, 0, stream>>>(attb, woutb, out);
}
